// SeeThroughMultiQ_35304631173914
// MI455X (gfx1250) — matrix operand order repaired and checked by reading, not compiled, not run on hardware
//
#include <hip/hip_runtime.h>

typedef __attribute__((ext_vector_type(16))) _Float16 v16h;
typedef __attribute__((ext_vector_type(8)))  float    v8f;

#define CDIM 256
#define KT 32
#define MT 32

// aMode: 0 = f16 buffer A0[row*256+k]
//        1 = f32 A0[row*256+k] + f32 A1[row*256+k]
//        2 = f32 A0[(row/49)*256+k] + f32 A1[(row%49)*256+k]   (q_split / q2)
//        4 = conv-z: f32 A0/A1 indexed [b,k,hw] with row = b*4096+hw
// epMode: 0 = f32 plain, 1 = f16 relu, 2 = f16 (acc*mulBuf), 3 = f16 sigmoid(LN),
//         4 = f32 LN, 5 = f16 plain.  All include +bias.
// W is pre-converted f16, row-major [k][n].
__global__ __launch_bounds__(256) void stq_gemm32(
    int aMode, const void* A0, const void* A1,
    const _Float16* W, const float* bias,
    int epMode, void* Out, const _Float16* mulBuf,
    const float* gamma, const float* beta)
{
  __shared__ _Float16 As[MT * KT];        // 32 rows x 32 k
  __shared__ _Float16 Wt[CDIM * KT];      // transposed: [n][k]
  __shared__ float lnbuf[MT * CDIM];      // 32 KB (LN modes)
  __shared__ float lmean[MT];
  __shared__ float lrstd[MT];

  const int tid  = threadIdx.x;
  const int lane = tid & 31;
  const int w    = tid >> 5;          // wave id 0..7 -> N columns [32w, 32w+32)
  const int hsel = lane >> 4;         // lane half
  const int nl   = lane & 15;
  const int row0 = blockIdx.x * MT;

  v8f acc00 = {}, acc01 = {}, acc10 = {}, acc11 = {};

  for (int k0 = 0; k0 < CDIM; k0 += KT) {
    __syncthreads();
    // ---- stage A tile (32x32 halves, 4 per thread) ----
    for (int e = tid; e < MT * KT; e += 256) {
      int r = e >> 5, kk = e & 31;
      int row = row0 + r;
      int k = k0 + kk;
      float val;
      if (aMode == 0) {
        val = (float)((const _Float16*)A0)[(size_t)row * CDIM + k];
      } else if (aMode == 1) {
        val = ((const float*)A0)[(size_t)row * CDIM + k] +
              ((const float*)A1)[(size_t)row * CDIM + k];
      } else if (aMode == 2) {
        val = ((const float*)A0)[(size_t)(row / 49) * CDIM + k] +
              ((const float*)A1)[(size_t)(row % 49) * CDIM + k];
      } else {  // conv-z: feat[b, k, hw], row = b*4096+hw
        int b = row >> 12, hw = row & 4095;
        size_t idx = ((size_t)b * CDIM + k) * 4096 + hw;
        val = ((const float*)A0)[idx] + ((const float*)A1)[idx];
      }
      As[e] = (_Float16)val;
    }
    // ---- stage W tile transposed (f16 direct): Wt[n][k] = W[k0+k][n] ----
    for (int e = tid; e < CDIM * KT; e += 256) {
      int n = e & 255, kk = e >> 8;
      Wt[n * KT + kk] = W[(size_t)(k0 + kk) * CDIM + n];
    }
    __syncthreads();

    // ---- build fragments per ISA 16-bit layouts ----
    v16h af0, af1, bf0, bf1;
    const _Float16* ap0 = &As[nl * KT + 8 * hsel];          // rows 0..15
    const _Float16* ap1 = &As[(16 + nl) * KT + 8 * hsel];   // rows 16..31
#pragma unroll
    for (int i = 0; i < 8; ++i) {
      af0[i] = ap0[i]; af0[8 + i] = ap0[16 + i];
      af1[i] = ap1[i]; af1[8 + i] = ap1[16 + i];
    }
    const _Float16* bp0 = &Wt[(w * 32 + nl) * KT + 8 * hsel];        // n tile 0
    const _Float16* bp1 = &Wt[(w * 32 + 16 + nl) * KT + 8 * hsel];   // n tile 1
#pragma unroll
    for (int i = 0; i < 8; ++i) { bf0[i] = bp0[i]; bf0[8 + i] = bp0[16 + i]; bf1[i] = bp1[i]; bf1[8 + i] = bp1[16 + i]; }

    acc00 = __builtin_amdgcn_wmma_f32_16x16x32_f16(false, af0, false, bf0, (short)0, acc00, false, false);
    acc01 = __builtin_amdgcn_wmma_f32_16x16x32_f16(false, af0, false, bf1, (short)0, acc01, false, false);
    acc10 = __builtin_amdgcn_wmma_f32_16x16x32_f16(false, af1, false, bf0, (short)0, acc10, false, false);
    acc11 = __builtin_amdgcn_wmma_f32_16x16x32_f16(false, af1, false, bf1, (short)0, acc11, false, false);
  }

  // ---- epilogue ----
  const int nbase = w * 32;
  const float b0 = bias[nbase + nl];
  const float b1 = bias[nbase + 16 + nl];
#pragma unroll
  for (int r = 0; r < 8; ++r) {
    int m0 = 8 * hsel + r;                // D layout: VGPR r -> M = r + 8*(lane>>4)
    int m1 = 16 + m0;
    float v00 = acc00[r] + b0;
    float v01 = acc01[r] + b1;
    float v10 = acc10[r] + b0;
    float v11 = acc11[r] + b1;
    size_t o00 = ((size_t)row0 + m0) * CDIM + nbase + nl;
    size_t o01 = o00 + 16;
    size_t o10 = ((size_t)row0 + m1) * CDIM + nbase + nl;
    size_t o11 = o10 + 16;
    if (epMode == 0) {
      ((float*)Out)[o00] = v00; ((float*)Out)[o01] = v01;
      ((float*)Out)[o10] = v10; ((float*)Out)[o11] = v11;
    } else if (epMode == 1) {
      ((_Float16*)Out)[o00] = (_Float16)fmaxf(v00, 0.f);
      ((_Float16*)Out)[o01] = (_Float16)fmaxf(v01, 0.f);
      ((_Float16*)Out)[o10] = (_Float16)fmaxf(v10, 0.f);
      ((_Float16*)Out)[o11] = (_Float16)fmaxf(v11, 0.f);
    } else if (epMode == 2) {
      ((_Float16*)Out)[o00] = (_Float16)(v00 * (float)mulBuf[o00]);
      ((_Float16*)Out)[o01] = (_Float16)(v01 * (float)mulBuf[o01]);
      ((_Float16*)Out)[o10] = (_Float16)(v10 * (float)mulBuf[o10]);
      ((_Float16*)Out)[o11] = (_Float16)(v11 * (float)mulBuf[o11]);
    } else if (epMode == 5) {
      ((_Float16*)Out)[o00] = (_Float16)v00;
      ((_Float16*)Out)[o01] = (_Float16)v01;
      ((_Float16*)Out)[o10] = (_Float16)v10;
      ((_Float16*)Out)[o11] = (_Float16)v11;
    } else {  // 3/4: stage for LayerNorm
      lnbuf[m0 * CDIM + nbase + nl] = v00;
      lnbuf[m0 * CDIM + nbase + 16 + nl] = v01;
      lnbuf[m1 * CDIM + nbase + nl] = v10;
      lnbuf[m1 * CDIM + nbase + 16 + nl] = v11;
    }
  }
  if (epMode == 3 || epMode == 4) {
    __syncthreads();
    if (tid < MT) {
      float s = 0.f, s2 = 0.f;
      for (int c = 0; c < CDIM; ++c) { float x = lnbuf[tid * CDIM + c]; s += x; s2 += x * x; }
      float mean = s * (1.f / CDIM);
      float var = s2 * (1.f / CDIM) - mean * mean;
      lmean[tid] = mean;
      lrstd[tid] = rsqrtf(var + 1e-5f);
    }
    __syncthreads();
    int m = tid >> 3;                  // 8 threads per row, 32 rows
    size_t row = (size_t)row0 + m;
    float mean = lmean[m], rstd = lrstd[m];
    for (int i = 0; i < 32; ++i) {
      int c = (tid & 7) + 8 * i;
      float y = (lnbuf[m * CDIM + c] - mean) * rstd * gamma[c] + beta[c];
      if (epMode == 3) {
        ((_Float16*)Out)[row * CDIM + c] = (_Float16)(1.f / (1.f + expf(-y)));
      } else {
        ((float*)Out)[row * CDIM + c] = y;
      }
    }
  }
}

// f32 [256,256] weight -> f16 copy (L2-resident, shared by all GEMM blocks)
__global__ __launch_bounds__(256) void stq_cvt_w(const float* W, _Float16* Wh) {
  int i = blockIdx.x * 256 + threadIdx.x;   // 65536 elements
  Wh[i] = (_Float16)W[i];
}

// feat [8,256,4096] f32 -> ft [8,4096,256] f16 (channel-last for sampling)
__global__ __launch_bounds__(256) void stq_transpose(const float* feat, _Float16* ft) {
  const int total = 8 * 256 * 4096;
  for (int i = blockIdx.x * 256 + threadIdx.x; i < total; i += gridDim.x * 256) {
    int b = i >> 20;
    int rem = i & ((1 << 20) - 1);
    int c = rem >> 12;
    int hw = rem & 4095;
    ft[(((size_t)b << 12) + hw) * 256 + c] = (_Float16)feat[i];
  }
}

// one block per (b,nq,g) row; thread = channel. Bilinear, align_corners=False, zero pad.
__global__ __launch_bounds__(256) void stq_sample(
    const float* bboxes, const _Float16* zt, const _Float16* ft,
    _Float16* roi, _Float16* vbuf)
{
  int rowM = blockIdx.x;
  int b = rowM / 4900;
  int rem = rowM % 4900;
  int nq = rem / 49;
  int g = rem % 49;
  int s1 = g / 7, s2 = g % 7;
  const float* bb = bboxes + ((size_t)b * 100 + nq) * 4;
  float bx = bb[0], by = bb[1], bh = bb[2], bw = bb[3];
  float h_sl = bh * (1.f / 7.f), w_sl = bw * (1.f / 7.f);
  float ys = h_sl * s1 + 0.5f * h_sl + by - 0.5f * bh;
  float xs = w_sl * s2 + 0.5f * w_sl + bx - 0.5f * bw;
  float gx = fminf(fmaxf(2.f * xs - 1.f, -1.f), 1.f);
  float gy = fminf(fmaxf(2.f * ys - 1.f, -1.f), 1.f);
  float ix = ((gx + 1.f) * 64.f - 1.f) * 0.5f;
  float iy = ((gy + 1.f) * 64.f - 1.f) * 0.5f;
  float x0f = floorf(ix), y0f = floorf(iy);
  float wx1 = ix - x0f, wx0 = 1.f - wx1;
  float wy1 = iy - y0f, wy0 = 1.f - wy1;
  int x0 = (int)x0f, y0 = (int)y0f, x1 = x0 + 1, y1 = y0 + 1;
  float wt[4] = {wx0 * wy0, wx1 * wy0, wx0 * wy1, wx1 * wy1};
  int xi[4] = {x0, x1, x0, x1};
  int yi[4] = {y0, y0, y1, y1};
  int idx[4];
#pragma unroll
  for (int t = 0; t < 4; ++t) {
    bool valid = (xi[t] >= 0) && (xi[t] < 64) && (yi[t] >= 0) && (yi[t] < 64);
    int xc = xi[t] < 0 ? 0 : (xi[t] > 63 ? 63 : xi[t]);
    int yc = yi[t] < 0 ? 0 : (yi[t] > 63 ? 63 : yi[t]);
    idx[t] = yc * 64 + xc;
    if (!valid) wt[t] = 0.f;
  }
  int c = threadIdx.x;
  size_t base = (size_t)b * 4096 * 256;
  float rv = 0.f, vv = 0.f;
#pragma unroll
  for (int t = 0; t < 4; ++t) {
    size_t o = base + (size_t)idx[t] * 256 + c;
    rv += wt[t] * (float)zt[o];
    vv += wt[t] * (float)ft[o];
  }
  size_t orow = (size_t)rowM * 256 + c;
  roi[orow] = (_Float16)rv;
  vbuf[orow] = (_Float16)vv;
}

__global__ __launch_bounds__(256) void stq_combine(
    const _Float16* g1, const _Float16* f1, const _Float16* g2, const _Float16* f2,
    _Float16* out, int total)
{
  for (int i = blockIdx.x * 256 + threadIdx.x; i < total; i += gridDim.x * 256)
    out[i] = (_Float16)((float)g1[i] * (float)f1[i] + (float)g2[i] * (float)f2[i]);
}

extern "C" void kernel_launch(void* const* d_in, const int* in_sizes, int n_in,
                              void* d_out, int out_size, void* d_ws, size_t ws_size,
                              hipStream_t stream) {
  (void)in_sizes; (void)n_in; (void)out_size; (void)ws_size;
  const float* q    = (const float*)d_in[0];
  const float* qpe  = (const float*)d_in[1];
  const float* feat = (const float*)d_in[2];
  const float* fpe  = (const float*)d_in[3];
  const float* bbox = (const float*)d_in[4];
  const float* ge   = (const float*)d_in[5];
  const float* bz   = (const float*)d_in[19];
  const float* bq   = (const float*)d_in[20];
  const float* b1l  = (const float*)d_in[21];
  const float* b2l  = (const float*)d_in[22];
  const float* g1b1 = (const float*)d_in[23];
  const float* g1b2 = (const float*)d_in[24];
  const float* g2b1 = (const float*)d_in[25];
  const float* g2b2 = (const float*)d_in[26];
  const float* f1b1 = (const float*)d_in[27];
  const float* f1b2 = (const float*)d_in[28];
  const float* f2b1 = (const float*)d_in[29];
  const float* f2b2 = (const float*)d_in[30];
  const float* outb = (const float*)d_in[31];
  const float* g1beta  = (const float*)d_in[32];
  const float* g2beta  = (const float*)d_in[33];
  const float* outbeta = (const float*)d_in[34];
  const float* g1g  = (const float*)d_in[35];
  const float* g2g  = (const float*)d_in[36];
  const float* outg = (const float*)d_in[37];

  char* ws = (char*)d_ws;
  const size_t MC = 39200ull * 256;   // 10,035,200
  const size_t WN = 65536;            // 256*256
  size_t off = 0;
  auto take = [&](size_t b) { size_t o = off; off += (b + 255) & ~size_t(255); return o; };
  size_t o_wh  = take(13 * WN * 2);   // 13 f16 weight matrices
  size_t o_qs  = take(800ull * 256 * 4);
  size_t o_zt  = take(32768ull * 256 * 2);
  size_t o_ft  = take(32768ull * 256 * 2);
  size_t o_roi = take(MC * 2);
  size_t o_v   = take(MC * 2);
  size_t o_t1  = take(MC * 2);
  size_t o_kk  = take(MC * 2);
  size_t o_h   = take(MC * 2);
  size_t o_g2  = take(MC * 2);

  _Float16* wh = (_Float16*)(ws + o_wh);
  // f16 weights, order: Wz Wq W1l W2l g1w1 g1w2 g2w1 g2w2 f1w1 f1w2 f2w1 f2w2 outw
  _Float16* whp[13];
  for (int i = 0; i < 13; ++i) {
    whp[i] = wh + (size_t)i * WN;
    stq_cvt_w<<<256, 256, 0, stream>>>((const float*)d_in[6 + i], whp[i]);
  }
  const _Float16 *hWz = whp[0], *hWq = whp[1], *hW1l = whp[2], *hW2l = whp[3],
                 *hg1w1 = whp[4], *hg1w2 = whp[5], *hg2w1 = whp[6], *hg2w2 = whp[7],
                 *hf1w1 = whp[8], *hf1w2 = whp[9], *hf2w1 = whp[10], *hf2w2 = whp[11],
                 *houtw = whp[12];

  float*     qs    = (float*)(ws + o_qs);
  _Float16*  zt    = (_Float16*)(ws + o_zt);
  _Float16*  ft    = (_Float16*)(ws + o_ft);
  _Float16*  roi   = (_Float16*)(ws + o_roi);
  _Float16*  vbuf  = (_Float16*)(ws + o_v);
  _Float16*  t1    = (_Float16*)(ws + o_t1);
  _Float16*  kk    = (_Float16*)(ws + o_kk);
  _Float16*  hb    = (_Float16*)(ws + o_h);
  _Float16*  gate2 = (_Float16*)(ws + o_g2);
  _Float16*  gate1 = t1;                 // t1 dead after step 6
  _Float16*  f1o   = zt;                 // zt/ft dead after sampling
  _Float16*  f2o   = kk;                 // kk dead after step 9
  _Float16*  aout  = roi;                // roi dead after step 6
  float*     outp  = (float*)d_out;

  // 1. qs = (q+qpe)@Wq + bq  (f32, M=800)
  stq_gemm32<<<800 / 32, 256, 0, stream>>>(1, q, qpe, hWq, bq, 0, qs, nullptr, nullptr, nullptr);
  // 2. zt[b,hw,c] = (feat+feat_pe)@Wz + bz  (f16, M=32768)
  stq_gemm32<<<32768 / 32, 256, 0, stream>>>(4, feat, fpe, hWz, bz, 5, zt, nullptr, nullptr, nullptr);
  // 3. feat -> channel-last f16
  stq_transpose<<<4096, 256, 0, stream>>>(feat, ft);
  // 4. ROI bilinear sample: roi (from z), vbuf (from feat)
  stq_sample<<<39200, 256, 0, stream>>>(bbox, zt, ft, roi, vbuf);
  // 5. t1 = q_split@W1l + b1l  (f16)
  stq_gemm32<<<1225, 256, 0, stream>>>(2, qs, ge, hW1l, b1l, 5, t1, nullptr, nullptr, nullptr);
  // 6. kk = (roi@W2l + b2l) * t1
  stq_gemm32<<<1225, 256, 0, stream>>>(0, roi, nullptr, hW2l, b2l, 2, kk, t1, nullptr, nullptr);
  // 7-8. gate1 = sigmoid(LN(relu(kk@g1w1+b)@g1w2+b))
  stq_gemm32<<<1225, 256, 0, stream>>>(0, kk, nullptr, hg1w1, g1b1, 1, hb, nullptr, nullptr, nullptr);
  stq_gemm32<<<1225, 256, 0, stream>>>(0, hb, nullptr, hg1w2, g1b2, 3, gate1, nullptr, g1g, g1beta);
  // 9-10. gate2
  stq_gemm32<<<1225, 256, 0, stream>>>(0, kk, nullptr, hg2w1, g2b1, 1, hb, nullptr, nullptr, nullptr);
  stq_gemm32<<<1225, 256, 0, stream>>>(0, hb, nullptr, hg2w2, g2b2, 3, gate2, nullptr, g2g, g2beta);
  // 11-12. f1 = mlp(q2)   (q2 = q + ge fused in A-load)
  stq_gemm32<<<1225, 256, 0, stream>>>(2, q, ge, hf1w1, f1b1, 1, hb, nullptr, nullptr, nullptr);
  stq_gemm32<<<1225, 256, 0, stream>>>(0, hb, nullptr, hf1w2, f1b2, 5, f1o, nullptr, nullptr, nullptr);
  // 13-14. f2 = mlp(v)
  stq_gemm32<<<1225, 256, 0, stream>>>(0, vbuf, nullptr, hf2w1, f2b1, 1, hb, nullptr, nullptr, nullptr);
  stq_gemm32<<<1225, 256, 0, stream>>>(0, hb, nullptr, hf2w2, f2b2, 5, f2o, nullptr, nullptr, nullptr);
  // 15. out = gate1*f1 + gate2*f2
  stq_combine<<<4096, 256, 0, stream>>>(gate1, f1o, gate2, f2o, aout, (int)MC);
  // 16. result = LN(out@out_w + out_b) -> d_out (f32)
  stq_gemm32<<<1225, 256, 0, stream>>>(0, aout, nullptr, houtw, outb, 4, outp, nullptr, outg, outbeta);
}
